// VisualMamba_34462817583261
// MI455X (gfx1250) — matrix operand order repaired and checked by reading, not compiled, not run on hardware
//
#include <hip/hip_runtime.h>

// ---------------------------------------------------------------------------
// Vision-Mamba forward for MI455X (gfx1250), wave32 + WMMA f32_16x16x32_f16.
// GEMM staging uses GLOBAL_LOAD_ASYNC_TO_LDS_B128 (ASYNCcnt-tracked).
// ---------------------------------------------------------------------------

typedef _Float16 half_t;
typedef _Float16 v8h  __attribute__((ext_vector_type(8)));
typedef _Float16 v16h __attribute__((ext_vector_type(16)));
typedef float    v8f  __attribute__((ext_vector_type(8)));
typedef int      v4i  __attribute__((ext_vector_type(4)));

#if __has_builtin(__builtin_amdgcn_global_load_async_to_lds_b128)
#define ASYNC_LDS 1
typedef __attribute__((address_space(1))) v4i* as1_v4i_p;
typedef __attribute__((address_space(3))) v4i* as3_v4i_p;
#define GLOBAL_V4(p) ((as1_v4i_p)(p))
#define LDS_V4(p)    ((as3_v4i_p)(p))
#endif

static constexpr int BB     = 64;
static constexpr int IMGS   = 224;
static constexpr int PS     = 16;
static constexpr int CINC   = 3;
static constexpr int DM     = 192;
static constexpr int DEPTH  = 24;
static constexpr int DSTATE = 16;
static constexpr int DINNER = 384;
static constexpr int DTR    = 12;
static constexpr int DCONV  = 4;
static constexpr int NTOK   = 196;            // (224/16)^2
static constexpr int MTOT   = BB * NTOK;      // 12544  (multiple of 128)
static constexpr int KPATCH = CINC * PS * PS; // 768
static constexpr int XZW    = 2 * DINNER;     // 768
static constexpr int PROJLD = 64;             // x_proj rows padded 44 -> 64

// ---------------------------------------------------------------------------
// Generic f32 -> f16 conversion
// ---------------------------------------------------------------------------
__global__ void cvt_f32_to_f16_kernel(const float* __restrict__ src,
                                      half_t* __restrict__ dst, int n) {
    int i = blockIdx.x * blockDim.x + threadIdx.x;
    if (i < n) dst[i] = (half_t)src[i];
}

// x_proj_w (DEPTH, 44, 384) -> padded f16 (DEPTH, 64, 384), rows 44..63 zero
__global__ void cvt_pad_xproj_kernel(const float* __restrict__ src,
                                     half_t* __restrict__ dst) {
    int i = blockIdx.x * blockDim.x + threadIdx.x;
    int n = DEPTH * PROJLD * DINNER;
    if (i >= n) return;
    int k = i % DINNER;
    int r = (i / DINNER) % PROJLD;
    int l = i / (DINNER * PROJLD);
    dst[i] = (r < (DTR + 2 * DSTATE))
                 ? (half_t)src[((size_t)l * 44 + r) * DINNER + k]
                 : (half_t)0.0f;
}

// ---------------------------------------------------------------------------
// im2col: img (B,3,224,224) -> A (12544 x 768) f16, row = token, col=(c,iy,ix)
// ---------------------------------------------------------------------------
__global__ void im2col_kernel(const float* __restrict__ img,
                              half_t* __restrict__ A) {
    int i = blockIdx.x * blockDim.x + threadIdx.x;
    int n = MTOT * KPATCH;
    if (i >= n) return;
    int col = i % KPATCH;
    int m   = i / KPATCH;
    int b   = m / NTOK;
    int pp  = m % NTOK;
    int py = pp / 14, px = pp % 14;
    int c  = col / (PS * PS);
    int rem = col % (PS * PS);
    int iy = rem / PS, ix = rem % PS;
    size_t src = (((size_t)b * CINC + c) * IMGS + (py * PS + iy)) * IMGS + (px * PS + ix);
    A[i] = (half_t)img[src];
}

// ---------------------------------------------------------------------------
// WMMA GEMM:  C(MxN) = A16(MxK) * W16(NxK)^T  (+bias), optional f16 mirror.
// Block: 128 threads (4 waves). Workgroup tile 128x64, wave tile 32x64
// (2 A-fragments x 4 B-fragments = 8 WMMA per 32-K step; B frag reused x2).
// M % 128 == 0, N % 64 == 0, K % 32 == 0 (guaranteed by padding).
// ---------------------------------------------------------------------------
__global__ __launch_bounds__(128)
void wmma_gemm_nt_kernel(const half_t* __restrict__ A,
                         const half_t* __restrict__ W,
                         const float* __restrict__ bias,
                         float* __restrict__ C,
                         half_t* __restrict__ C16,
                         int K, int N) {
    __shared__ __align__(16) half_t sA[128 * 40]; // 128 rows, 32 halves + pad
    __shared__ __align__(16) half_t sB[64 * 40];

    const int tid  = threadIdx.x;
    const int m0   = blockIdx.x * 128;
    const int n0   = blockIdx.y * 64;
    const int lane = tid & 31;
    const int wv   = tid >> 5;           // wave id 0..3 -> 32-row M strip
    const int r    = lane & 15;
    const int hi   = (lane < 16) ? 0 : 1;
    const int aoff = hi ? 8 : 0;         // A frag K-half select
    const int boff = hi ? 8 : 0;         // B frag K-half select

    v8f acc[2][4] = {};

    for (int kk = 0; kk < K; kk += 32) {
        // ---- stage 128x32 A tile and 64x32 W tile into LDS ----
#pragma unroll
        for (int c = 0; c < 4; ++c) {    // A: 512 chunks of 8 halves
            int ch = tid + c * 128;
            int row = ch >> 2, col = (ch & 3) << 3;
            const half_t* gp = A + (size_t)(m0 + row) * K + kk + col;
#ifdef ASYNC_LDS
            __builtin_amdgcn_global_load_async_to_lds_b128(
                GLOBAL_V4(gp), LDS_V4(&sA[row * 40 + col]), 0, 0);
#else
            *(v8h*)&sA[row * 40 + col] = *(const v8h*)gp;
#endif
        }
#pragma unroll
        for (int c = 0; c < 2; ++c) {    // B: 256 chunks of 8 halves
            int ch = tid + c * 128;
            int row = ch >> 2, col = (ch & 3) << 3;
            const half_t* gp = W + (size_t)(n0 + row) * K + kk + col;
#ifdef ASYNC_LDS
            __builtin_amdgcn_global_load_async_to_lds_b128(
                GLOBAL_V4(gp), LDS_V4(&sB[row * 40 + col]), 0, 0);
#else
            *(v8h*)&sB[row * 40 + col] = *(const v8h*)gp;
#endif
        }
        if (kk + 32 < K) {   // speculative prefetch of next K tile into L2
            __builtin_prefetch(A + (size_t)(m0 + tid) * K + kk + 32, 0, 0);
            __builtin_prefetch(W + (size_t)(n0 + (tid & 63)) * K + kk + 32, 0, 0);
        }
#ifdef ASYNC_LDS
#if __has_builtin(__builtin_amdgcn_s_wait_asynccnt)
        __builtin_amdgcn_s_wait_asynccnt(0);
#else
        asm volatile("s_wait_asynccnt 0" ::: "memory");
#endif
#endif
        __syncthreads();

        // ---- A fragments: lane holds row m=r, K pattern {c..c+7, 16+c..} ----
        v16h av[2];
#pragma unroll
        for (int i = 0; i < 2; ++i) {
            v8h alo = *(const v8h*)&sA[(wv * 32 + i * 16 + r) * 40 + aoff];
            v8h ahi = *(const v8h*)&sA[(wv * 32 + i * 16 + r) * 40 + 16 + aoff];
            av[i] = __builtin_shufflevector(alo, ahi, 0, 1, 2, 3, 4, 5, 6, 7,
                                            8, 9, 10, 11, 12, 13, 14, 15);
        }
#pragma unroll
        for (int j = 0; j < 4; ++j) {
            // B fragment: lane holds col n=r, K pattern {c..c+7, 16+c..} from W row n
            v8h blo = *(const v8h*)&sB[(j * 16 + r) * 40 + boff];
            v8h bhi = *(const v8h*)&sB[(j * 16 + r) * 40 + 16 + boff];
            v16h bv = __builtin_shufflevector(blo, bhi, 0, 1, 2, 3, 4, 5, 6, 7,
                                              8, 9, 10, 11, 12, 13, 14, 15);
            acc[0][j] = __builtin_amdgcn_wmma_f32_16x16x32_f16(
                false, av[0], false, bv, (short)0, acc[0][j], false, false);
            acc[1][j] = __builtin_amdgcn_wmma_f32_16x16x32_f16(
                false, av[1], false, bv, (short)0, acc[1][j], false, false);
        }
        __syncthreads();
    }

    // ---- store: VGPR rr -> M = rr (+8 for upper lanes), N = lane%16 ----
#pragma unroll
    for (int i = 0; i < 2; ++i) {
#pragma unroll
        for (int j = 0; j < 4; ++j) {
            int ncol = n0 + j * 16 + r;
            float badd = bias ? bias[ncol] : 0.0f;
#pragma unroll
            for (int rr = 0; rr < 8; ++rr) {
                int mrow = m0 + wv * 32 + i * 16 + (hi ? rr + 8 : rr);
                float v = acc[i][j][rr] + badd;
                C[(size_t)mrow * N + ncol] = v;
                if (C16) C16[(size_t)mrow * N + ncol] = (half_t)v;
            }
        }
    }
}

// ---------------------------------------------------------------------------
// Depthwise causal conv (4 taps along L) + bias + SiLU; dual f32/f16 output
// ---------------------------------------------------------------------------
__global__ void conv_silu_kernel(const float* __restrict__ xz,
                                 const float* __restrict__ cw,
                                 const float* __restrict__ cb,
                                 float* __restrict__ xc32,
                                 half_t* __restrict__ xc16) {
    int i = blockIdx.x * blockDim.x + threadIdx.x;
    if (i >= MTOT * DINNER) return;
    int d = i % DINNER;
    int m = i / DINNER;
    int t = m % NTOK;
    float acc = cb[d];
#pragma unroll
    for (int k = 0; k < DCONV; ++k) {
        int tt = t - (DCONV - 1) + k;
        if (tt >= 0)
            acc += xz[(size_t)(m - (DCONV - 1) + k) * XZW + d] * cw[d * DCONV + k];
    }
    float s = acc / (1.0f + __expf(-acc));   // silu
    xc32[i] = s;
    xc16[i] = (half_t)s;
}

// ---------------------------------------------------------------------------
// dt = softplus(proj[:, :12] @ dt_w^T + dt_b)
// ---------------------------------------------------------------------------
__global__ void dt_softplus_kernel(const float* __restrict__ proj,
                                   const float* __restrict__ dtw,
                                   const float* __restrict__ dtb,
                                   float* __restrict__ dt32) {
    int i = blockIdx.x * blockDim.x + threadIdx.x;
    if (i >= MTOT * DINNER) return;
    int d = i % DINNER;
    int m = i / DINNER;
    float acc = dtb[d];
#pragma unroll
    for (int rj = 0; rj < DTR; ++rj)
        acc += proj[(size_t)m * PROJLD + rj] * dtw[d * DTR + rj];
    dt32[i] = (acc > 20.0f) ? acc : log1pf(__expf(acc));
}

// ---------------------------------------------------------------------------
// Selective scan: one block per batch, thread = channel d, h[16] in registers,
// B_t/C_t broadcast through LDS each step. Then gate with silu(z) -> y16.
// ---------------------------------------------------------------------------
__global__ __launch_bounds__(DINNER)
void scan_kernel(const float* __restrict__ proj,
                 const float* __restrict__ dt32,
                 const float* __restrict__ xc32,
                 const float* __restrict__ xz32,
                 const float* __restrict__ Alog,
                 const float* __restrict__ Dp,
                 half_t* __restrict__ y16) {
    __shared__ float sBt[DSTATE];
    __shared__ float sCt[DSTATE];
    const int b = blockIdx.x;
    const int d = threadIdx.x;

    float a[DSTATE];
#pragma unroll
    for (int s = 0; s < DSTATE; ++s) a[s] = -__expf(Alog[d * DSTATE + s]);
    float h[DSTATE];
#pragma unroll
    for (int s = 0; s < DSTATE; ++s) h[s] = 0.0f;
    const float Dv = Dp[d];

    for (int t = 0; t < NTOK; ++t) {
        const int m = b * NTOK + t;
        if (d < 2 * DSTATE) {
            int s = d & (DSTATE - 1);
            float v = proj[(size_t)m * PROJLD + DTR + (d < DSTATE ? 0 : DSTATE) + s];
            if (d < DSTATE) sBt[s] = v; else sCt[s] = v;
        }
        __syncthreads();
        const float dtv = dt32[(size_t)m * DINNER + d];
        const float u   = xc32[(size_t)m * DINNER + d];
        const float du  = dtv * u;
        float y = 0.0f;
#pragma unroll
        for (int s = 0; s < DSTATE; ++s) {
            h[s] = h[s] * __expf(dtv * a[s]) + du * sBt[s];
            y += h[s] * sCt[s];
        }
        float yv = y + u * Dv;
        float z = xz32[(size_t)m * XZW + DINNER + d];
        yv *= z / (1.0f + __expf(-z));       // * silu(z)
        y16[(size_t)m * DINNER + d] = (half_t)yv;
        __syncthreads();
    }
}

// ---------------------------------------------------------------------------
// LayerNorm stats: one wave per token (wave32), 192 values per token
// ---------------------------------------------------------------------------
__global__ __launch_bounds__(128)
void ln_stats_kernel(const float* __restrict__ x32,
                     float* __restrict__ mu, float* __restrict__ rstd) {
    int token = blockIdx.x * 4 + (threadIdx.x >> 5);
    int lane  = threadIdx.x & 31;
    float s1 = 0.0f, s2 = 0.0f;
    for (int i = lane; i < DM; i += 32) {
        float v = x32[(size_t)token * DM + i];
        s1 += v; s2 += v * v;
    }
#pragma unroll
    for (int off = 16; off > 0; off >>= 1) {
        s1 += __shfl_xor(s1, off, 32);
        s2 += __shfl_xor(s2, off, 32);
    }
    if (lane == 0) {
        float m = s1 / DM;
        float var = s2 / DM - m * m;
        mu[token] = m;
        rstd[token] = rsqrtf(var + 1e-5f);
    }
}

// ---------------------------------------------------------------------------
// out[b,d] = mean_t( (x - mu)*rstd ) * nw[d] + nb[d]   (deterministic)
// ---------------------------------------------------------------------------
__global__ void final_mean_kernel(const float* __restrict__ x32,
                                  const float* __restrict__ mu,
                                  const float* __restrict__ rstd,
                                  const float* __restrict__ nw,
                                  const float* __restrict__ nb,
                                  float* __restrict__ out) {
    int i = blockIdx.x * blockDim.x + threadIdx.x;
    if (i >= BB * DM) return;
    int d = i % DM;
    int b = i / DM;
    float acc = 0.0f;
    for (int t = 0; t < NTOK; ++t) {
        int m = b * NTOK + t;
        acc += (x32[(size_t)m * DM + d] - mu[m]) * rstd[m];
    }
    out[i] = (acc / NTOK) * nw[d] + nb[d];
}

// ---------------------------------------------------------------------------
// Host-side orchestration
// ---------------------------------------------------------------------------
extern "C" void kernel_launch(void* const* d_in, const int* in_sizes, int n_in,
                              void* d_out, int out_size, void* d_ws, size_t ws_size,
                              hipStream_t stream) {
    (void)in_sizes; (void)n_in; (void)out_size; (void)ws_size;

    const float* img        = (const float*)d_in[0];
    const float* patch_w    = (const float*)d_in[1];
    const float* patch_b    = (const float*)d_in[2];
    const float* in_proj_w  = (const float*)d_in[3];
    const float* conv_w     = (const float*)d_in[4];
    const float* conv_b     = (const float*)d_in[5];
    const float* x_proj_w   = (const float*)d_in[6];
    const float* dt_w       = (const float*)d_in[7];
    const float* dt_b       = (const float*)d_in[8];
    const float* A_log      = (const float*)d_in[9];
    const float* D_param    = (const float*)d_in[10];
    const float* out_proj_w = (const float*)d_in[11];
    const float* norm_w     = (const float*)d_in[12];
    const float* norm_b     = (const float*)d_in[13];

    // --- workspace carve-out (256B aligned) ---
    char* wsb = (char*)d_ws;
    size_t off = 0;
    auto carve = [&](size_t bytes) -> char* {
        char* p = wsb + off;
        off = (off + bytes + 255) & ~(size_t)255;
        return p;
    };
    half_t* w16_patch = (half_t*)carve((size_t)DM * KPATCH * 2);
    half_t* w16_in    = (half_t*)carve((size_t)DEPTH * XZW * DM * 2);
    half_t* w16_xp    = (half_t*)carve((size_t)DEPTH * PROJLD * DINNER * 2);
    half_t* w16_out   = (half_t*)carve((size_t)DEPTH * DM * DINNER * 2);
    float*  xz32      = (float*)carve((size_t)MTOT * XZW * 4);   // 38.5 MB
    half_t* im2col16  = (half_t*)xz32;    // alias: im2col dead before layer 0
    float*  x32       = (float*)carve((size_t)MTOT * DM * 4);
    half_t* x16       = (half_t*)carve((size_t)MTOT * DM * 2);
    float*  xc32      = (float*)carve((size_t)MTOT * DINNER * 4);
    half_t* xc16      = (half_t*)carve((size_t)MTOT * DINNER * 2);
    float*  proj32    = (float*)carve((size_t)MTOT * PROJLD * 4);
    float*  dt32      = (float*)carve((size_t)MTOT * DINNER * 4);
    half_t* y16       = (half_t*)carve((size_t)MTOT * DINNER * 2);
    float*  muv       = (float*)carve((size_t)MTOT * 4);
    float*  rstdv     = (float*)carve((size_t)MTOT * 4);

    auto g1 = [](int n) { return (n + 255) / 256; };

    // --- weight conversion (once per call) ---
    cvt_f32_to_f16_kernel<<<g1(DM * KPATCH), 256, 0, stream>>>(patch_w, w16_patch, DM * KPATCH);
    cvt_f32_to_f16_kernel<<<g1(DEPTH * XZW * DM), 256, 0, stream>>>(in_proj_w, w16_in, DEPTH * XZW * DM);
    cvt_f32_to_f16_kernel<<<g1(DEPTH * DM * DINNER), 256, 0, stream>>>(out_proj_w, w16_out, DEPTH * DM * DINNER);
    cvt_pad_xproj_kernel<<<g1(DEPTH * PROJLD * DINNER), 256, 0, stream>>>(x_proj_w, w16_xp);

    // --- patch embedding: im2col + WMMA GEMM (+bias) ---
    im2col_kernel<<<g1(MTOT * KPATCH), 256, 0, stream>>>(img, im2col16);
    wmma_gemm_nt_kernel<<<dim3(MTOT / 128, DM / 64), 128, 0, stream>>>(
        im2col16, w16_patch, patch_b, x32, x16, KPATCH, DM);

    // --- 24 Mamba layers ---
    for (int l = 0; l < DEPTH; ++l) {
        const half_t* wl_in  = w16_in  + (size_t)l * XZW * DM;
        const half_t* wl_xp  = w16_xp  + (size_t)l * PROJLD * DINNER;
        const half_t* wl_out = w16_out + (size_t)l * DM * DINNER;
        const float*  cw     = conv_w  + (size_t)l * DINNER * DCONV;
        const float*  cb     = conv_b  + (size_t)l * DINNER;
        const float*  dtwl   = dt_w    + (size_t)l * DINNER * DTR;
        const float*  dtbl   = dt_b    + (size_t)l * DINNER;
        const float*  Alogl  = A_log   + (size_t)l * DINNER * DSTATE;
        const float*  Dl     = D_param + (size_t)l * DINNER;

        // xz = x @ in_w^T : (12544 x 768)
        wmma_gemm_nt_kernel<<<dim3(MTOT / 128, XZW / 64), 128, 0, stream>>>(
            x16, wl_in, nullptr, xz32, nullptr, DM, XZW);
        // causal depthwise conv + silu
        conv_silu_kernel<<<g1(MTOT * DINNER), 256, 0, stream>>>(xz32, cw, cb, xc32, xc16);
        // proj = xc @ xpw^T : (12544 x 64 padded)
        wmma_gemm_nt_kernel<<<dim3(MTOT / 128, PROJLD / 64), 128, 0, stream>>>(
            xc16, wl_xp, nullptr, proj32, nullptr, DINNER, PROJLD);
        // dt = softplus(proj[:,:12] @ dtw^T + dtb)
        dt_softplus_kernel<<<g1(MTOT * DINNER), 256, 0, stream>>>(proj32, dtwl, dtbl, dt32);
        // selective scan + D skip + silu(z) gate -> y16
        scan_kernel<<<BB, DINNER, 0, stream>>>(proj32, dt32, xc32, xz32, Alogl, Dl, y16);
        // x = y @ outw^T : (12544 x 192), mirrored to f16 for next layer
        wmma_gemm_nt_kernel<<<dim3(MTOT / 128, DM / 64), 128, 0, stream>>>(
            y16, wl_out, nullptr, x32, x16, DINNER, DM);
    }

    // --- LayerNorm + mean over tokens ---
    ln_stats_kernel<<<MTOT / 4, 128, 0, stream>>>(x32, muv, rstdv);
    final_mean_kernel<<<g1(BB * DM), 256, 0, stream>>>(x32, muv, rstdv, norm_w, norm_b,
                                                       (float*)d_out);
}
